// RPN_67800353734942
// MI455X (gfx1250) — matrix operand order repaired and checked by reading, compiled, not run on hardware
//
#include <hip/hip_runtime.h>

// ---------------------------------------------------------------------------
// RPN for MI455X (gfx1250).  Dominant cost: 3x3 conv 512->512 over 8x64x64,
// ~155 GFLOP, L2-resident -> compute bound -> bf16 WMMA (v_wmma_f32_16x16x32_bf16).
// Weights double-buffered into LDS via the Tensor Data Mover (TENSORcnt).
// ---------------------------------------------------------------------------

typedef __attribute__((ext_vector_type(8)))  float   v8f;
typedef __attribute__((ext_vector_type(8)))  __bf16  v8bf;
typedef __attribute__((ext_vector_type(16))) __bf16  v16bf;
typedef __attribute__((ext_vector_type(4)))  unsigned u32x4;
typedef __attribute__((ext_vector_type(8)))  int      i32x8;
typedef __attribute__((ext_vector_type(4)))  int      i32x4;

#if defined(__AMDGCN__) && __has_builtin(__builtin_amdgcn_tensor_load_to_lds)
#define HAS_TDM 1
#else
#define HAS_TDM 0
#endif

static __device__ __forceinline__ void wait_tensor0() {
#if defined(__AMDGCN__)
#if __has_builtin(__builtin_amdgcn_s_wait_tensorcnt)
  __builtin_amdgcn_s_wait_tensorcnt(0);
#else
  asm volatile("s_wait_tensorcnt 0x0" ::: "memory");
#endif
#endif
}

// ------------------------- constants -----------------------------------
#define HH 64
#define WW 64
#define CIN 512
#define COUT 512
#define NA 9
#define NANCH 36864            // 64*64*9
#define PRE_TOPN 20
#define AFTER_TOPN 10

// 9 anchors from _gen_anchors(16,(0.5,1,2),(8,16,32)) (row = ratio-major)
__constant__ float ANC[9][4] = {
  {-84.f,  -40.f,  99.f,  55.f},
  {-176.f, -88.f,  191.f, 103.f},
  {-360.f, -184.f, 375.f, 199.f},
  {-56.f,  -56.f,  71.f,  71.f},
  {-120.f, -120.f, 135.f, 135.f},
  {-248.f, -248.f, 263.f, 263.f},
  {-36.f,  -80.f,  51.f,  95.f},
  {-80.f,  -168.f, 95.f,  183.f},
  {-168.f, -344.f, 183.f, 359.f}};

// ------------------- kernel 1: NCHW f32 -> NHWC bf16 --------------------
__global__ __launch_bounds__(256) void k_nchw2nhwc(const float* __restrict__ in,
                                                   __bf16* __restrict__ out) {
  __shared__ float tile[32][33];
  const int b  = blockIdx.z;
  const int s0 = blockIdx.x * 32;          // spatial (y*64+x)
  const int c0 = blockIdx.y * 32;          // channel
  const int tx = threadIdx.x & 31, ty = threadIdx.x >> 5;   // 32 x 8
#pragma unroll
  for (int r = 0; r < 32; r += 8)
    tile[ty + r][tx] = in[(((size_t)b * CIN + c0 + ty + r) << 12) + s0 + tx];
  __syncthreads();
#pragma unroll
  for (int r = 0; r < 32; r += 8)
    out[((size_t)b * 4096 + s0 + ty + r) * CIN + c0 + tx] = (__bf16)tile[tx][ty + r];
}

// --------------- kernel 2: conv_w OIHW f32 -> [tap][o][c] bf16 ----------
__global__ __launch_bounds__(256) void k_wprep(const float* __restrict__ cw,
                                               __bf16* __restrict__ w2) {
  size_t i = (size_t)blockIdx.x * 256 + threadIdx.x;   // (t*512+o)*512+c
  if (i < (size_t)9 * 512 * 512) {
    int c = (int)(i & 511);
    size_t r = i >> 9;
    int o = (int)(r & 511);
    int t = (int)(r >> 9);
    int ky = t / 3, kx = t % 3;
    w2[i] = (__bf16)cw[(((size_t)o * 512 + c) * 3 + ky) * 3 + kx];
  }
}

// ----------------------- TDM weight-tile load ---------------------------
#if HAS_TDM
static __device__ __forceinline__ void tdm_wtile(const __bf16* w2, unsigned ldsAddr,
                                                 int t, int osb, int c0) {
  unsigned long long g =
      (unsigned long long)(size_t)(w2 + ((size_t)(t * 512 + osb)) * 512 + c0);
  u32x4 g0;
  g0[0] = 1u;                               // count=1, user descriptor
  g0[1] = ldsAddr;                          // LDS byte address
  g0[2] = (unsigned)g;                      // global addr lo
  g0[3] = (unsigned)((g >> 32) & 0x1FFFFFFu) | (2u << 30);  // addr hi | type=2
  i32x8 g1;
  g1[0] = 1 << 16;                          // data_size = 2 bytes, no mask/pads
  g1[1] = (int)(512u << 16);                // tensor_dim0[15:0]=512
  g1[2] = (int)(512u << 16);                // tensor_dim0 hi=0 | tensor_dim1 lo=512
  g1[3] = (int)(32u << 16);                 // tensor_dim1 hi=0 | tile_dim0=32 (c)
  g1[4] = 128;                              // tile_dim1=128 (o), tile_dim2=0
  g1[5] = 512;                              // tensor_dim0_stride = 512
  g1[6] = 0;
  g1[7] = 0;
  i32x4 z4; z4[0] = z4[1] = z4[2] = z4[3] = 0;
  i32x8 z8;
#pragma unroll
  for (int k = 0; k < 8; ++k) z8[k] = 0;
  // 6-arg toolchain form: (g0, g1, g2, g3, g2b, cpol)
  __builtin_amdgcn_tensor_load_to_lds(g0, g1, z4, z4, z8, 0);
}
#endif

// --------- kernel 3: 3x3 conv + bias + ReLU via bf16 WMMA ---------------
// GEMM: D[o][x] = sum_k W[o][k] * F[k][x];  M=128 o-chs (this block),
// N=64 x positions (one row y), K = 9 taps * 512 ch in chunks of 32.
__global__ __launch_bounds__(256) void k_conv3x3(const __bf16* __restrict__ fe,
                                                 const __bf16* __restrict__ w2,
                                                 const float* __restrict__ conv_b,
                                                 __bf16* __restrict__ fconv) {
  __shared__ __bf16 frow[66 * 256];          // halo row window, 256 chans staged
  __shared__ __bf16 wbuf[2][128 * 32];       // double-buffered weight tile
  const int tid = threadIdx.x;
  const int osb = blockIdx.x * 128;
  const int y   = blockIdx.y;
  const int b   = blockIdx.z;
  const int lane = tid & 31, wv = tid >> 5;
  const int lm = lane & 15, h = lane >> 4;

  v8f acc[4];
#pragma unroll
  for (int j = 0; j < 4; ++j)
#pragma unroll
    for (int k = 0; k < 8; ++k) acc[j][k] = 0.0f;

  // ---- weight chunk loader: it -> (tap, c0) ----
  auto wload = [&](int it, int buf) {
    const int ky = it / 48, half = (it / 24) & 1, kx = (it >> 3) % 3, cc = it & 7;
    const int t = ky * 3 + kx;
    const int c0 = half * 256 + cc * 32;
#if HAS_TDM
    if (tid < 32) {                          // one wave issues the DMA
      unsigned lds = (unsigned)(size_t)(void*)&wbuf[buf][0];
      tdm_wtile(w2, lds, t, osb, c0);
    }
#else
    const __bf16* src = w2 + ((size_t)(t * 512 + osb)) * 512 + c0;
    for (int i = tid; i < 512; i += 256) {   // 128 rows x 4 v8 chunks
      int o = i >> 2, c8 = (i & 3) << 3;
      *(v8bf*)&wbuf[buf][o * 32 + c8] = *(const v8bf*)(src + (size_t)o * 512 + c8);
    }
#endif
  };

  wload(0, 0);                               // prologue

#pragma unroll 2
  for (int it = 0; it < 144; ++it) {
    const int ky = it / 48, half = (it / 24) & 1, kx = (it >> 3) % 3, cc = it & 7;
    if ((it % 24) == 0) {                    // restage feature halo row
      __syncthreads();
      const int yy = y + ky - 1;
      const bool rowOK = (yy >= 0) && (yy < HH);
      const __bf16* src =
          fe + (((size_t)b * HH + (rowOK ? yy : 0)) * WW) * CIN + half * 256;
      for (int i = tid; i < 66 * 32; i += 256) {
        int p = i >> 5, c8 = (i & 31) << 3, xx = p - 1;
        v8bf v;
        if (rowOK && xx >= 0 && xx < WW) {
          v = *(const v8bf*)(src + (size_t)xx * CIN + c8);
        } else {
#pragma unroll
          for (int k = 0; k < 8; ++k) v[k] = (__bf16)0.0f;
        }
        *(v8bf*)&frow[(p << 8) + c8] = v;
      }
    }
#if HAS_TDM
    if (tid < 32) wait_tensor0();            // current buffer's DMA done
#endif
    __syncthreads();                         // publish wbuf/frow to all waves
    if (it + 1 < 144) wload(it + 1, (it + 1) & 1);

    // ---- fetch ALL fragments first (one dscnt wait), then WMMA chain ----
    const int bi = it & 1;
    const __bf16* wr = &wbuf[bi][((wv << 4) + lm) * 32 + (h << 3)];
    v8bf a0 = *(const v8bf*)wr;              // K = 8h .. 8h+7
    v8bf a1 = *(const v8bf*)(wr + 16);       // K = 16+8h .. 16+8h+7
    v8bf b0[4], b1[4];
    const __bf16* fb0 = &frow[((lm + kx) << 8) + (cc << 5) + (h << 3)];
#pragma unroll
    for (int j = 0; j < 4; ++j) {
      const __bf16* fb = fb0 + (j << 12);    // +16 positions = 4096 elems
      b0[j] = *(const v8bf*)fb;              // K = 8h .. 8h+7 of column x
      b1[j] = *(const v8bf*)(fb + 16);       // K = 16+8h .. 16+8h+7
    }
    v16bf Af = __builtin_shufflevector(a0, a1, 0, 1, 2, 3, 4, 5, 6, 7,
                                       8, 9, 10, 11, 12, 13, 14, 15);
#pragma unroll
    for (int j = 0; j < 4; ++j) {
      v16bf Bf = __builtin_shufflevector(b0[j], b1[j], 0, 1, 2, 3, 4, 5, 6, 7,
                                         8, 9, 10, 11, 12, 13, 14, 15);
      acc[j] = __builtin_amdgcn_wmma_f32_16x16x32_bf16(false, Af, false, Bf,
                                                       (short)0, acc[j], false, false);
    }
  }

  // ---- epilogue: bias + ReLU + store NHWC bf16 ----
  const int obase = osb + (wv << 4);
#pragma unroll
  for (int j = 0; j < 4; ++j) {
    const int x = (j << 4) + lm;
    const size_t ob = ((size_t)(b * HH + y) * WW + x) * COUT;
#pragma unroll
    for (int r = 0; r < 8; ++r) {
      const int o = obase + r + (h << 3);
      float v = acc[j][r] + conv_b[o];
      fconv[ob + o] = (__bf16)(v > 0.0f ? v : 0.0f);
    }
  }
}

// -------- kernel 4: 1x1 heads + softmax + decode + clip + score ---------
__global__ __launch_bounds__(256) void k_heads(const __bf16* __restrict__ fconv,
                                               const float* __restrict__ cls_w,
                                               const float* __restrict__ cls_b,
                                               const float* __restrict__ reg_w,
                                               const float* __restrict__ reg_b,
                                               const float* __restrict__ im_info,
                                               float* __restrict__ scores,
                                               float* __restrict__ boxes) {
  __shared__ __bf16 fv[16][512];
  __shared__ float lg[16][64];
  const int tid = threadIdx.x;
  const size_t P0 = (size_t)blockIdx.x * 16;     // 16 positions per block
  for (int i = tid; i < 1024; i += 256) {        // 16*512 bf16, v8 chunks
    int pl = i >> 6, c8 = (i & 63) << 3;
    *(v8bf*)&fv[pl][c8] = *(const v8bf*)(fconv + (P0 + pl) * 512 + c8);
  }
  __syncthreads();
  const int ch = tid & 63, pl0 = tid >> 6;
  if (ch < 54) {
    const float* wr = (ch < 18) ? (cls_w + (size_t)ch * 512) : (reg_w + (size_t)(ch - 18) * 512);
    const float bias = (ch < 18) ? cls_b[ch] : reg_b[ch - 18];
    for (int pl = pl0; pl < 16; pl += 4) {
      float s = 0.0f;
      for (int c8 = 0; c8 < 512; c8 += 8) {
        v8bf f8 = *(const v8bf*)&fv[pl][c8];
#pragma unroll
        for (int k = 0; k < 8; ++k) s += (float)f8[k] * wr[c8 + k];
      }
      lg[pl][ch] = s + bias;
    }
  }
  __syncthreads();
  if (tid < 144) {
    const int pl = tid / 9, a = tid % 9;
    const size_t P = P0 + pl;
    const int b = (int)(P >> 12);
    const int s_ = (int)(P & 4095), y = s_ >> 6, x = s_ & 63;
    const float fg = 1.0f / (1.0f + __expf(lg[pl][a] - lg[pl][9 + a]));
    const float dx = lg[pl][18 + a * 4 + 0], dy = lg[pl][18 + a * 4 + 1];
    const float dw = lg[pl][18 + a * 4 + 2], dh = lg[pl][18 + a * 4 + 3];
    const float ax1 = ANC[a][0] + x * 16.0f, ay1 = ANC[a][1] + y * 16.0f;
    const float ax2 = ANC[a][2] + x * 16.0f, ay2 = ANC[a][3] + y * 16.0f;
    const float wA = ax2 - ax1 + 1.0f, hA = ay2 - ay1 + 1.0f;
    const float cx = ax1 + 0.5f * (wA - 1.0f), cy = ay1 + 0.5f * (hA - 1.0f);
    const float pcx = dx * wA + cx, pcy = dy * hA + cy;
    const float pw = __expf(dw) * wA, ph = __expf(dh) * hA;
    const float hmax = im_info[b * 3 + 0] - 1.0f;
    const float wmax = im_info[b * 3 + 1] - 1.0f;
    const float msz = 5.0f * im_info[b * 3 + 2];
    float x1 = pcx - 0.5f * pw, y1 = pcy - 0.5f * ph;
    float x2 = pcx + 0.5f * pw, y2 = pcy + 0.5f * ph;
    x1 = fminf(fmaxf(x1, 0.0f), wmax); y1 = fminf(fmaxf(y1, 0.0f), hmax);
    x2 = fminf(fmaxf(x2, 0.0f), wmax); y2 = fminf(fmaxf(y2, 0.0f), hmax);
    const bool valid = (x2 - x1 + 1.0f >= msz) && (y2 - y1 + 1.0f >= msz);
    const int n = (y * 64 + x) * NA + a;
    scores[(size_t)b * NANCH + n] = valid ? fg : -__builtin_inff();
    float* bp = boxes + ((size_t)b * NANCH + n) * 4;
    bp[0] = x1; bp[1] = y1; bp[2] = x2; bp[3] = y2;
  }
}

// ------------- kernel 5: per-batch top-20 + NMS + output ----------------
__global__ __launch_bounds__(256) void k_proposals(float* __restrict__ scores,
                                                   const float* __restrict__ boxes,
                                                   float* __restrict__ out) {
  __shared__ float rv[256];
  __shared__ int ri[256];
  __shared__ float tb[PRE_TOPN][4];
  const int b = blockIdx.x, tid = threadIdx.x;
  float* sc = scores + (size_t)b * NANCH;
  for (int i = 0; i < PRE_TOPN; ++i) {
    float best = -__builtin_inff(); int bi = 0x7FFFFFFF;
    for (int n = tid; n < NANCH; n += 256) {
      float v = sc[n];
      if (v > best || (v == best && n < bi)) { best = v; bi = n; }
    }
    rv[tid] = best; ri[tid] = bi;
    __syncthreads();
    for (int s = 128; s > 0; s >>= 1) {
      if (tid < s) {
        if (rv[tid + s] > rv[tid] || (rv[tid + s] == rv[tid] && ri[tid + s] < ri[tid])) {
          rv[tid] = rv[tid + s]; ri[tid] = ri[tid + s];
        }
      }
      __syncthreads();
    }
    if (tid == 0) {
      int n = ri[0];
      sc[n] = -__builtin_inff();             // regenerated each call by k_heads
      const float* bp = boxes + ((size_t)b * NANCH + n) * 4;
      tb[i][0] = bp[0]; tb[i][1] = bp[1]; tb[i][2] = bp[2]; tb[i][3] = bp[3];
    }
    __syncthreads();
  }
  if (tid == 0) {
    bool keep[PRE_TOPN];
    float area[PRE_TOPN];
    for (int i = 0; i < PRE_TOPN; ++i) {
      keep[i] = true;
      area[i] = (tb[i][2] - tb[i][0] + 1.0f) * (tb[i][3] - tb[i][1] + 1.0f);
    }
    for (int i = 0; i < PRE_TOPN; ++i) {
      if (!keep[i]) continue;
      for (int j = i + 1; j < PRE_TOPN; ++j) {
        float xx1 = fmaxf(tb[i][0], tb[j][0]);
        float yy1 = fmaxf(tb[i][1], tb[j][1]);
        float xx2 = fminf(tb[i][2], tb[j][2]);
        float yy2 = fminf(tb[i][3], tb[j][3]);
        float iw = fmaxf(xx2 - xx1 + 1.0f, 0.0f);
        float ih = fmaxf(yy2 - yy1 + 1.0f, 0.0f);
        float inter = iw * ih;
        float iou = inter / (area[i] + area[j] - inter);
        if (iou > 0.7f) keep[j] = false;
      }
    }
    int ord[PRE_TOPN]; int slot = 0;
    for (int i = 0; i < PRE_TOPN; ++i) if (keep[i])  ord[slot++] = i;   // stable
    for (int i = 0; i < PRE_TOPN; ++i) if (!keep[i]) ord[slot++] = i;
    for (int s = 0; s < AFTER_TOPN; ++s) {
      int o = ord[s];
      bool v = keep[o];
      float* op = out + ((size_t)b * AFTER_TOPN + s) * 5;
      op[0] = (float)b;
      op[1] = v ? tb[o][0] : 0.0f;
      op[2] = v ? tb[o][1] : 0.0f;
      op[3] = v ? tb[o][2] : 0.0f;
      op[4] = v ? tb[o][3] : 0.0f;
    }
  }
}

// ---------------------------------------------------------------------------
extern "C" void kernel_launch(void* const* d_in, const int* in_sizes, int n_in,
                              void* d_out, int out_size, void* d_ws, size_t ws_size,
                              hipStream_t stream) {
  (void)in_sizes; (void)n_in; (void)out_size; (void)ws_size;
  const float* feature = (const float*)d_in[0];
  const float* im_info = (const float*)d_in[1];
  const float* conv_w  = (const float*)d_in[2];
  const float* conv_b  = (const float*)d_in[3];
  const float* cls_w   = (const float*)d_in[4];
  const float* cls_b   = (const float*)d_in[5];
  const float* reg_w   = (const float*)d_in[6];
  const float* reg_b   = (const float*)d_in[7];

  // workspace layout (needs ~77.8 MB)
  char* ws = (char*)d_ws;
  __bf16* fe    = (__bf16*)(ws);                    // 33,554,432 B  NHWC bf16
  __bf16* w2    = (__bf16*)(ws + 33554432);         //  4,718,592 B  [tap][o][c]
  __bf16* fconv = (__bf16*)(ws + 38273024);         // 33,554,432 B  NHWC bf16
  float*  scores = (float*)(ws + 71827456);         //  1,179,648 B
  float*  boxes  = (float*)(ws + 73007104);         //  4,718,592 B
  float*  out = (float*)d_out;                      // (8,10,5) f32

  k_nchw2nhwc<<<dim3(128, 16, 8), 256, 0, stream>>>(feature, fe);
  k_wprep<<<dim3(9216), 256, 0, stream>>>(conv_w, w2);
  k_conv3x3<<<dim3(4, 64, 8), 256, 0, stream>>>(fe, w2, conv_b, fconv);
  k_heads<<<dim3(2048), 256, 0, stream>>>(fconv, cls_w, cls_b, reg_w, reg_b,
                                          im_info, scores, boxes);
  k_proposals<<<dim3(8), 256, 0, stream>>>(scores, boxes, out);
}
